// HFFVSSBlock_12498354831586
// MI455X (gfx1250) — matrix operand order repaired and checked by reading, not compiled, not run on hardware
//
#include <hip/hip_runtime.h>
#include <hip/hip_bf16.h>
#include <math.h>

typedef __attribute__((ext_vector_type(16))) __bf16 v16bf;
typedef __attribute__((ext_vector_type(8)))  float  v8f;

#define DIM    256
#define INNER  384
#define DSTATE 8
#define DTRANK 48
#define NTOK   16384   /* 4*64*64 */
#define EPSLN  1e-5f

// LDS tile strides (in bf16 elements), padded to kill bank conflicts
#define APAD 40   /* 32 + 8 */
#define BPAD 40

__device__ __forceinline__ float sigmoidf_(float x) { return 1.f / (1.f + __expf(-x)); }
__device__ __forceinline__ float geluf_(float x)    { return 0.5f * x * (1.f + erff(x * 0.70710678118654752f)); }

__device__ __forceinline__ unsigned short bf16b(float x) {
    return __builtin_bit_cast(unsigned short, (__bf16)x);
}
__device__ __forceinline__ unsigned pk_bf16(float lo, float hi) {
    return ((unsigned)bf16b(hi) << 16) | (unsigned)bf16b(lo);
}
__device__ __forceinline__ v16bf ld_frag16(const unsigned short* p0, const unsigned short* p1) {
    struct Q2 { uint4 a; uint4 b; };
    Q2 q;
    q.a = *(const uint4*)p0;   // 16B: 8 bf16, contiguous K
    q.b = *(const uint4*)p1;   // 16B: 8 bf16, contiguous K
    return __builtin_bit_cast(v16bf, q);
}

// ---------------------------------------------------------------------------
// LDS-staged WMMA GEMM: C[M,N] = epi(A[M,K](lda) * B[K,N] + bias) + res
// Requirements: M % 64 == 0, N % 128 == 0, K % 32 == 0, lda % 4 == 0.
// Block = 256 threads = 8 waves (4 in M x 2 in N). Block tile 64x128,
// each wave: 16 rows x 64 cols = 4 wmma tiles. K-step = 32.
// epi: 0 = none, 1 = silu, 2 = gelu.  bias/res may be null.
// ---------------------------------------------------------------------------
__global__ __launch_bounds__(256) void wmma_gemm_kernel(
    const float* __restrict__ A, const float* __restrict__ B,
    float* __restrict__ C, int M, int N, int K, int lda,
    const float* __restrict__ bias, const float* __restrict__ res, int epi)
{
    __shared__ alignas(16) unsigned short sA[64 * APAD];    //  5.0 KB
    __shared__ alignas(16) unsigned short sB[128 * BPAD];   // 10.0 KB

    const int tid  = threadIdx.x;
    const int lane = tid & 31;
    const int wave = tid >> 5;      // 0..7
    const int wm   = wave >> 1;     // 0..3  (M direction)
    const int wn   = wave & 1;      // 0..1  (N direction)
    const int half = lane >> 4;     // 0/1
    const int l16  = lane & 15;

    const int rowBase = blockIdx.y * 64;
    const int colBase = blockIdx.x * 128;

    // staging coordinates (no bounds checks needed by construction)
    const int arow = tid >> 2;            // 0..63
    const int akq  = (tid & 3) * 8;       // 0,8,16,24
    const float* aSrc = A + (size_t)(rowBase + arow) * (size_t)lda + akq;

    v8f acc[4];
#pragma unroll
    for (int j = 0; j < 4; ++j) acc[j] = (v8f){0.f,0.f,0.f,0.f,0.f,0.f,0.f,0.f};

    for (int kb = 0; kb < K; kb += 32) {
        __syncthreads();
        // ---- stage A tile: 64 x 32 f32 -> bf16 row-major in LDS ----
        {
            const float4* p = (const float4*)(aSrc + kb);
            float4 f0 = p[0];
            float4 f1 = p[1];
            uint4 u;
            u.x = pk_bf16(f0.x, f0.y);
            u.y = pk_bf16(f0.z, f0.w);
            u.z = pk_bf16(f1.x, f1.y);
            u.w = pk_bf16(f1.z, f1.w);
            *(uint4*)&sA[arow * APAD + akq] = u;
        }
        // ---- stage B tile: 32 x 128 f32 -> bf16 TRANSPOSED (col-major) ----
#pragma unroll
        for (int i = 0; i < 4; ++i) {
            int f4  = tid + i * 256;      // 0..1023 float4 chunks
            int br  = f4 >> 5;            // 0..31  (k row)
            int bc4 = f4 & 31;            // col/4
            float4 v = *(const float4*)&B[(size_t)(kb + br) * (size_t)N + colBase + bc4 * 4];
            sB[(bc4 * 4 + 0) * BPAD + br] = bf16b(v.x);
            sB[(bc4 * 4 + 1) * BPAD + br] = bf16b(v.y);
            sB[(bc4 * 4 + 2) * BPAD + br] = bf16b(v.z);
            sB[(bc4 * 4 + 3) * BPAD + br] = bf16b(v.w);
        }
        // prefetch next K-step (global_prefetch_b8)
        if (kb + 32 < K) {
            __builtin_prefetch(aSrc + kb + 32, 0, 1);
            __builtin_prefetch(&B[(size_t)(kb + 32 + (tid >> 3)) * (size_t)N + colBase + (tid & 7) * 16], 0, 1);
        }
        __syncthreads();

        // ---- A fragment: row = wm*16 + l16; K chunks {half*8..+7} and {16+half*8..+7}
        const unsigned short* ar = &sA[(wm * 16 + l16) * APAD];
        v16bf afrag = ld_frag16(ar + half * 8, ar + 16 + half * 8);

        // ---- 4 N-tiles: B fragment col = wn*64 + j*16 + l16; K chunks {half*8..+7} and {16+half*8..+7}
#pragma unroll
        for (int j = 0; j < 4; ++j) {
            const unsigned short* br = &sB[(wn * 64 + j * 16 + l16) * BPAD];
            v16bf bfrag = ld_frag16(br + half * 8, br + 16 + half * 8);
            acc[j] = __builtin_amdgcn_wmma_f32_16x16x32_bf16(false, afrag, false, bfrag,
                                                             (short)0, acc[j], false, false);
        }
    }

    // ---- epilogue: C layout VGPR i -> m = i + 8*half, n = lane&15 ----
#pragma unroll
    for (int j = 0; j < 4; ++j) {
        int n = colBase + wn * 64 + j * 16 + l16;
#pragma unroll
        for (int i = 0; i < 8; ++i) {
            int m = rowBase + wm * 16 + i + half * 8;
            size_t o = (size_t)m * (size_t)N + (size_t)n;
            float v = acc[j][i];
            if (bias) v += bias[n];
            if (epi == 1)      v = v * sigmoidf_(v);
            else if (epi == 2) v = geluf_(v);
            if (res) v += res[o];
            C[o] = v;
        }
    }
}

// ---------------------------------------------------------------------------
// Weight prep: concat fwd/bwd xproj [384,64]+[384,64] -> [384,128]
// ---------------------------------------------------------------------------
__global__ __launch_bounds__(256) void concat_xproj_kernel(
    const float* __restrict__ wf, const float* __restrict__ wb, float* __restrict__ dst)
{
    int id = blockIdx.x * 256 + threadIdx.x;     // 384*128
    int k = id >> 7, c = id & 127;
    dst[id] = (c < 64) ? wf[k * 64 + c] : wb[k * 64 + (c - 64)];
}

// Weight prep: zero-pad dt_W [48,384] -> [64,384]
__global__ __launch_bounds__(256) void pad_dtw_kernel(
    const float* __restrict__ src, float* __restrict__ dst)
{
    int id = blockIdx.x * 256 + threadIdx.x;     // 64*384
    int k = id / 384;
    dst[id] = (k < DTRANK) ? src[id] : 0.f;
}

// ---------------------------------------------------------------------------
// LayerNorm over last dim = 256.  One wave per token, wave32 shfl reductions.
// ---------------------------------------------------------------------------
__global__ __launch_bounds__(256) void layernorm256_kernel(
    const float* __restrict__ x, const float* __restrict__ g,
    const float* __restrict__ b, float* __restrict__ y)
{
    const int lane = threadIdx.x & 31;
    const int wave = threadIdx.x >> 5;
    const int tok  = blockIdx.x * 8 + wave;
    const float* xr = x + (size_t)tok * DIM;
    float* yr = y + (size_t)tok * DIM;

    float v[8];
    float s = 0.f;
#pragma unroll
    for (int i = 0; i < 8; ++i) { v[i] = xr[i * 32 + lane]; s += v[i]; }
#pragma unroll
    for (int off = 16; off > 0; off >>= 1) s += __shfl_xor(s, off, 32);
    float mu = s * (1.f / 256.f);
    float q = 0.f;
#pragma unroll
    for (int i = 0; i < 8; ++i) { float d = v[i] - mu; q += d * d; }
#pragma unroll
    for (int off = 16; off > 0; off >>= 1) q += __shfl_xor(q, off, 32);
    float rinv = rsqrtf(q * (1.f / 256.f) + EPSLN);
#pragma unroll
    for (int i = 0; i < 8; ++i) {
        int c = i * 32 + lane;
        yr[c] = (v[i] - mu) * rinv * g[c] + b[c];
    }
}

// ---------------------------------------------------------------------------
// Bidirectional 4-token windowed S6 scan.  One thread per (window, channel).
// projfb: [NTOK,128], fwd cols 0..63, bwd cols 64..127.
// ---------------------------------------------------------------------------
__global__ __launch_bounds__(256) void s6_scan_kernel(
    const float* __restrict__ xz,      // [NTOK, 768], x_proj = cols 0..383
    const float* __restrict__ projfb,  // [NTOK, 128]
    const float* __restrict__ dtf,     // [NTOK, 384]
    const float* __restrict__ dtb,     // [NTOK, 384]
    const float* __restrict__ fAlog, const float* __restrict__ fD,
    const float* __restrict__ bAlog, const float* __restrict__ bD,
    float* __restrict__ yssm)          // [NTOK, 384]
{
    int id = blockIdx.x * blockDim.x + threadIdx.x;   // 4096 * 384
    int win = id / INNER;
    int ch  = id - win * INNER;
    int bb  = win >> 10;
    int rem = win & 1023;
    int h2  = rem >> 5;
    int w2  = rem & 31;

    int tok[4];
#pragma unroll
    for (int t = 0; t < 4; ++t) {
        int dh = t >> 1, dw = t & 1;
        tok[t] = ((bb * 64 + h2 * 2 + dh) * 64 + (w2 * 2 + dw));
    }

    float xv[4], dfv[4], dbv[4];
#pragma unroll
    for (int t = 0; t < 4; ++t) {
        xv[t]  = xz[(size_t)tok[t] * 768 + ch];
        dfv[t] = dtf[(size_t)tok[t] * INNER + ch];
        dbv[t] = dtb[(size_t)tok[t] * INNER + ch];
    }

    float af[DSTATE], ab[DSTATE];
#pragma unroll
    for (int n = 0; n < DSTATE; ++n) {
        af[n] = -__expf(fAlog[ch * DSTATE + n]);
        ab[n] = -__expf(bAlog[ch * DSTATE + n]);
    }

    float yf[4], yb[4];
    {   // forward scan t = 0..3
        float st[DSTATE] = {0,0,0,0,0,0,0,0};
#pragma unroll
        for (int t = 0; t < 4; ++t) {
            const float* p = projfb + (size_t)tok[t] * 128;
            float d = dfv[t], y = 0.f;
#pragma unroll
            for (int n = 0; n < DSTATE; ++n) {
                float dA = __expf(d * af[n]);
                st[n] = dA * st[n] + d * p[DTRANK + n] * xv[t];
                y += st[n] * p[DTRANK + DSTATE + n];
            }
            yf[t] = y + xv[t] * fD[ch];
        }
    }
    {   // backward scan t = 3..0
        float st[DSTATE] = {0,0,0,0,0,0,0,0};
#pragma unroll
        for (int tt = 0; tt < 4; ++tt) {
            int t = 3 - tt;
            const float* p = projfb + (size_t)tok[t] * 128 + 64;
            float d = dbv[t], y = 0.f;
#pragma unroll
            for (int n = 0; n < DSTATE; ++n) {
                float dA = __expf(d * ab[n]);
                st[n] = dA * st[n] + d * p[DTRANK + n] * xv[t];
                y += st[n] * p[DTRANK + DSTATE + n];
            }
            yb[t] = y + xv[t] * bD[ch];
        }
    }
#pragma unroll
    for (int t = 0; t < 4; ++t)
        yssm[(size_t)tok[t] * INNER + ch] = 0.5f * (yf[t] + yb[t]);
}

// ---------------------------------------------------------------------------
// Depthwise 3x3 conv (SAME, zero pad) on x_proj (cols 0..383 of xz).
// ---------------------------------------------------------------------------
__global__ __launch_bounds__(256) void dwconv3_kernel(
    const float* __restrict__ xz, const float* __restrict__ kern,
    float* __restrict__ ydw)
{
    int id = blockIdx.x * blockDim.x + threadIdx.x;   // NTOK * 384
    int tok = id / INNER;
    int ch  = id - tok * INNER;
    int bb  = tok >> 12;
    int hw  = tok & 4095;
    int h   = hw >> 6;
    int w   = hw & 63;

    float acc = 0.f;
#pragma unroll
    for (int kh = 0; kh < 3; ++kh) {
        int hh = h + kh - 1;
        if (hh < 0 || hh > 63) continue;
#pragma unroll
        for (int kw = 0; kw < 3; ++kw) {
            int ww = w + kw - 1;
            if (ww < 0 || ww > 63) continue;
            int t2 = (bb << 12) + (hh << 6) + ww;
            acc += xz[(size_t)t2 * 768 + ch] * kern[(kh * 3 + kw) * INNER + ch];
        }
    }
    ydw[(size_t)tok * INNER + ch] = acc;
}

// ---------------------------------------------------------------------------
// Gated mix: ymix = (sigmoid(g)*yssm + (1-sigmoid(g))*yconv) * silu(z)
// ---------------------------------------------------------------------------
__global__ __launch_bounds__(256) void mix_kernel(
    const float* __restrict__ yssm, const float* __restrict__ yconv,
    const float* __restrict__ xz, const float* __restrict__ gate,
    float* __restrict__ ymix)
{
    int id = blockIdx.x * blockDim.x + threadIdx.x;   // NTOK * 384
    int tok = id / INNER;
    int ch  = id - tok * INNER;
    float alpha = sigmoidf_(gate[ch]);
    float y = alpha * yssm[id] + (1.f - alpha) * yconv[id];
    float z = xz[(size_t)tok * 768 + INNER + ch];
    ymix[id] = y * (z * sigmoidf_(z));
}

// ---------------------------------------------------------------------------
extern "C" void kernel_launch(void* const* d_in, const int* in_sizes, int n_in,
                              void* d_out, int out_size, void* d_ws, size_t ws_size,
                              hipStream_t stream)
{
    const float* x        = (const float*)d_in[0];
    const float* norm_g   = (const float*)d_in[1];
    const float* norm_b   = (const float*)d_in[2];
    const float* in_projW = (const float*)d_in[3];
    const float* f_xprojW = (const float*)d_in[4];
    const float* f_dtW    = (const float*)d_in[5];
    const float* f_dtbias = (const float*)d_in[6];
    const float* f_Alog   = (const float*)d_in[7];
    const float* f_D      = (const float*)d_in[8];
    const float* b_xprojW = (const float*)d_in[9];
    const float* b_dtW    = (const float*)d_in[10];
    const float* b_dtbias = (const float*)d_in[11];
    const float* b_Alog   = (const float*)d_in[12];
    const float* b_D      = (const float*)d_in[13];
    const float* dw_k     = (const float*)d_in[14];
    const float* pw_W     = (const float*)d_in[15];
    const float* gate     = (const float*)d_in[16];
    const float* out_W    = (const float*)d_in[17];
    const float* norm2_g  = (const float*)d_in[18];
    const float* norm2_b  = (const float*)d_in[19];
    const float* fuse_W1  = (const float*)d_in[20];
    const float* fuse_b1  = (const float*)d_in[21];
    const float* fuse_W2  = (const float*)d_in[22];
    const float* fuse_b2  = (const float*)d_in[23];
    float* out = (float*)d_out;

    // ---- workspace layout (floats) ----
    float* ws = (float*)d_ws;
    const size_t NT = NTOK;
    float* w_xn     = ws;                          // NT*256
    float* w_xz     = w_xn     + NT * 256;         // NT*768
    float* w_projfb = w_xz     + NT * 768;         // NT*128
    float* w_dtf    = w_projfb + NT * 128;         // NT*384
    float* w_dtb    = w_dtf    + NT * 384;         // NT*384
    float* w_yssm   = w_dtb    + NT * 384;         // NT*384
    float* w_ydw    = w_yssm   + NT * 384;         // NT*384
    float* w_xout   = w_ydw    + NT * 384;         // NT*256
    float* w_hid    = w_xout   + NT * 256;         // NT*512
    float* w_xprojc = w_hid    + NT * 512;         // 384*128
    float* w_dtWfp  = w_xprojc + 384 * 128;        // 64*384
    float* w_dtWbp  = w_dtWfp  + 64 * 384;         // 64*384
    float* w_yconv  = w_dtb;                       // reuse (consumed by scan)
    float* w_ymix   = w_dtf;                       // reuse (consumed by scan)
    float* w_h2     = w_xn;                        // reuse (consumed by GEMM1)

    const int M = NTOK;
    dim3 blk(256);

    // 0) weight prep: concat xproj, pad dt_W with zero rows 48..63
    concat_xproj_kernel<<<dim3((384 * 128) / 256), blk, 0, stream>>>(f_xprojW, b_xprojW, w_xprojc);
    pad_dtw_kernel<<<dim3((64 * 384) / 256), blk, 0, stream>>>(f_dtW, w_dtWfp);
    pad_dtw_kernel<<<dim3((64 * 384) / 256), blk, 0, stream>>>(b_dtW, w_dtWbp);

    // 1) LN1
    layernorm256_kernel<<<dim3(NTOK / 8), blk, 0, stream>>>(x, norm_g, norm_b, w_xn);

    // 2) in_proj: [M,256] x [256,768] -> xz
    wmma_gemm_kernel<<<dim3(768 / 128, M / 64), blk, 0, stream>>>(
        w_xn, in_projW, w_xz, M, 768, 256, 256, nullptr, nullptr, 0);

    // 3) xproj fwd+bwd fused: x_proj (lda=768 slice of xz) x [384,128] -> projfb
    wmma_gemm_kernel<<<dim3(1, M / 64), blk, 0, stream>>>(
        w_xz, w_xprojc, w_projfb, M, 128, 384, 768, nullptr, nullptr, 0);

    // 4) dt fwd/bwd: proj (lda=128, K=64 with zero-padded B) x [64,384] + bias, silu
    wmma_gemm_kernel<<<dim3(384 / 128, M / 64), blk, 0, stream>>>(
        w_projfb, w_dtWfp, w_dtf, M, 384, 64, 128, f_dtbias, nullptr, 1);
    wmma_gemm_kernel<<<dim3(384 / 128, M / 64), blk, 0, stream>>>(
        w_projfb + 64, w_dtWbp, w_dtb, M, 384, 64, 128, b_dtbias, nullptr, 1);

    // 5) bidirectional S6 scan -> yssm
    s6_scan_kernel<<<dim3((4096 * INNER) / 256), blk, 0, stream>>>(
        w_xz, w_projfb, w_dtf, w_dtb, f_Alog, f_D, b_Alog, b_D, w_yssm);

    // 6) depthwise 3x3 conv -> ydw
    dwconv3_kernel<<<dim3((NTOK * INNER) / 256), blk, 0, stream>>>(w_xz, dw_k, w_ydw);

    // 7) pointwise: ydw x [384,384], gelu -> yconv
    wmma_gemm_kernel<<<dim3(384 / 128, M / 64), blk, 0, stream>>>(
        w_ydw, pw_W, w_yconv, M, 384, 384, 384, nullptr, nullptr, 2);

    // 8) gated mix * silu(z) -> ymix
    mix_kernel<<<dim3((NTOK * INNER) / 256), blk, 0, stream>>>(
        w_yssm, w_yconv, w_xz, gate, w_ymix);

    // 9) out_proj + residual(x) -> xout
    wmma_gemm_kernel<<<dim3(256 / 128, M / 64), blk, 0, stream>>>(
        w_ymix, out_W, w_xout, M, 256, 384, 384, nullptr, x, 0);

    // 10) LN2 -> h2
    layernorm256_kernel<<<dim3(NTOK / 8), blk, 0, stream>>>(w_xout, norm2_g, norm2_b, w_h2);

    // 11) fuse1: h2 x [256,512] + b1, gelu -> hidden
    wmma_gemm_kernel<<<dim3(512 / 128, M / 64), blk, 0, stream>>>(
        w_h2, fuse_W1, w_hid, M, 512, 256, 256, fuse_b1, nullptr, 2);

    // 12) fuse2: hidden x [512,256] + b2 + residual(xout) -> out
    wmma_gemm_kernel<<<dim3(256 / 128, M / 64), blk, 0, stream>>>(
        w_hid, fuse_W2, out, M, 256, 512, 512, fuse_b2, w_xout, 0);

    (void)in_sizes; (void)n_in; (void)out_size; (void)ws_size;
}
